// WindowAttention_39135742002039
// MI455X (gfx1250) — matrix operand order repaired and checked by reading, not compiled, not run on hardware
//
#include <hip/hip_runtime.h>
#include <hip/hip_bf16.h>

#define WIN_N   49
#define NDIM    192
#define NHEADS  6
#define HDIM    32
#define NWIN    1024
#define NPAD    64
#define NBATCH  8192
#define SCALE_Q 0.17677669529663687f   /* 32^-0.5 */

typedef __bf16 bf16_t;
typedef __attribute__((ext_vector_type(16))) __bf16 v16bf;
typedef __attribute__((ext_vector_type(8)))  __bf16 v8bf;
typedef __attribute__((ext_vector_type(4)))  __bf16 v4bf;
typedef __attribute__((ext_vector_type(8)))  float  v8f;
typedef __attribute__((ext_vector_type(4)))  float  v4f;

__device__ __forceinline__ v8f wmma_bf16(v16bf a, v16bf b, v8f c) {
  // D = A(16x32) * B(32x16) + C, f32 accumulate
  return __builtin_amdgcn_wmma_f32_16x16x32_bf16(false, a, false, b, (short)0, c,
                                                 false, false);
}

// ---- A-tile (16x32, bf16) from row-major bf16 src with leading dim ld ----
// ISA 7.12.2: lanes 0-15: M=0..15, K={0..7,16..23}; lanes 16-31: same M, K={8..15,24..31}
__device__ __forceinline__ v16bf load_a16(const bf16_t* base, int row0, int ld,
                                          int kbase, int lane) {
  const int m  = lane & 15;
  const int k0 = kbase + ((lane & 16) ? 8 : 0);
  const bf16_t* p = base + (row0 + m) * ld + k0;
  v8bf lo = *(const v8bf*)(p);
  v8bf hi = *(const v8bf*)(p + 16);
  return __builtin_shufflevector(lo, hi, 0, 1, 2, 3, 4, 5, 6, 7,
                                          8, 9, 10, 11, 12, 13, 14, 15);
}

// ---- B-tile (32x16, bf16): src indexed src[n][k] (column n contiguous in K) ----
// lanes 0-15: col n, K={0..7,16..23}; lanes 16-31: col n, K={8..15,24..31}
__device__ __forceinline__ v16bf load_b16(const bf16_t* base, int col0, int ld,
                                          int kbase, int lane) {
  const int n  = lane & 15;
  const int k0 = kbase + ((lane & 16) ? 8 : 0);
  const bf16_t* p = base + (col0 + n) * ld + k0;
  v8bf lo = *(const v8bf*)(p);
  v8bf hi = *(const v8bf*)(p + 16);
  return __builtin_shufflevector(lo, hi, 0, 1, 2, 3, 4, 5, 6, 7,
                                          8, 9, 10, 11, 12, 13, 14, 15);
}

// ---------------- prep: weights -> bf16, gather rel-pos bias ----------------
__global__ void prep_kernel(const float* __restrict__ qkv_w,
                            const float* __restrict__ proj_w,
                            const float* __restrict__ rpb,
                            const int* __restrict__ rel_index,
                            bf16_t* __restrict__ qkvw, bf16_t* __restrict__ projw,
                            float* __restrict__ biasf) {
  const int tid = blockIdx.x * blockDim.x + threadIdx.x;
  if (tid < 3 * NDIM * NDIM) qkvw[tid] = (bf16_t)qkv_w[tid];
  if (tid < NDIM * NDIM) projw[tid] = (bf16_t)proj_w[tid];
  if (tid < NHEADS * WIN_N * WIN_N) {
    const int h  = tid / (WIN_N * WIN_N);
    const int nm = tid % (WIN_N * WIN_N);
    biasf[tid] = rpb[rel_index[nm] * NHEADS + h];
  }
}

// ---------------- fused window attention: 1 block == 1 window ----------------
struct __align__(64) SmemT {
  bf16_t xb[NPAD][NDIM];    // 24 KB  x window, bf16, zero-padded rows
  bf16_t q[NPAD][HDIM];     //  4 KB  (pre-scaled by 32^-0.5)
  bf16_t k[NPAD][HDIM];     //  4 KB
  bf16_t vT[HDIM][NPAD];    //  4 KB  (v transposed: [d][m])
  float  attn[NPAD][NPAD];  // 16 KB
  bf16_t p[NPAD][NPAD];     //  8 KB  (softmax probs, bf16)
  bf16_t o[NPAD][NDIM];     // 24 KB  (attention output, all heads)
};                          // total 84 KB (gfx1250: 320 KB/WG addressable)

// QKV epilogue: route one 16x16 accumulator tile to q / k / vT(LDS-transposed)
__device__ __forceinline__ void store_qkv_tile(SmemT& sm, const v8f& acc, int t,
                                               int h, int mt, int crow, int ccol,
                                               const float* __restrict__ qkv_b) {
  const int s    = t >> 1;                 // 0=q 1=k 2=v
  const int lc   = (t & 1) * 16 + ccol;    // local col within 32-wide head dim
  const float bi = qkv_b[s * NDIM + h * HDIM + (t & 1) * 16 + ccol];
#pragma unroll
  for (int i = 0; i < 8; ++i) {
    const int r = mt * 16 + crow + i;
    const float v = acc[i] + bi;
    if (s == 0)      sm.q[r][lc]  = (bf16_t)(v * SCALE_Q);
    else if (s == 1) sm.k[r][lc]  = (bf16_t)v;
    else             sm.vT[lc][r] = (bf16_t)v;
  }
}

__global__ void __launch_bounds__(256)
win_attn_kernel(const float* __restrict__ x, const float* __restrict__ mask,
                const float* __restrict__ qkv_b, const float* __restrict__ proj_b,
                const bf16_t* __restrict__ qkvw, const bf16_t* __restrict__ projw,
                const float* __restrict__ biasf, float* __restrict__ out) {
  __shared__ SmemT sm;
  const int b    = blockIdx.x;
  const int tid  = threadIdx.x;
  const int wave = __builtin_amdgcn_readfirstlane(tid >> 5);  // wave-uniform
  const int lane = tid & 31;
  const int win  = b & (NWIN - 1);
  const float* xw = x + (size_t)b * WIN_N * NDIM;
  const float* mw = mask + (size_t)win * WIN_N * WIN_N;

  const int crow = (lane >> 4) << 3;  // C/D layout: vgpr i -> row crow+i
  const int ccol = lane & 15;         // col = lane % 16

  // ---- stage x window into LDS as bf16 (once; reused by all 6 heads) ----
#pragma unroll 1
  for (int q4 = tid; q4 < (NPAD * NDIM) / 4; q4 += 256) {
    const int e0 = q4 * 4;
    const int r  = e0 / NDIM;
    v4f xv = {};
    if (r < WIN_N) xv = *(const v4f*)(xw + e0);
    v4bf bv;
#pragma unroll
    for (int i = 0; i < 4; ++i) bv[i] = (bf16_t)xv[i];
    *(v4bf*)(&sm.xb[0][0] + e0) = bv;
  }
  __syncthreads();

  for (int h = 0; h < NHEADS; ++h) {
    // ---- QKV GEMM for head h: 8 jobs = 4 mtiles x 2 halves, 3 acc each ----
    // head h owns 6 column-tiles t=0..5 at global col (t>>1)*192 + h*32 + (t&1)*16
    {
      const int mt   = wave >> 1;
      const int half = wave & 1;
      const int t0 = half * 3, t1 = t0 + 1, t2 = t0 + 2;
      const int g0 = (t0 >> 1) * NDIM + h * HDIM + (t0 & 1) * 16;
      const int g1 = (t1 >> 1) * NDIM + h * HDIM + (t1 & 1) * 16;
      const int g2 = (t2 >> 1) * NDIM + h * HDIM + (t2 & 1) * 16;
      v8f acc0 = {}, acc1 = {}, acc2 = {};
#pragma unroll
      for (int kk = 0; kk < 6; ++kk) {
        v16bf a  = load_a16(&sm.xb[0][0], mt * 16, NDIM, kk * 32, lane);
        v16bf b0 = load_b16(qkvw, g0, NDIM, kk * 32, lane);
        v16bf b1 = load_b16(qkvw, g1, NDIM, kk * 32, lane);
        v16bf b2 = load_b16(qkvw, g2, NDIM, kk * 32, lane);
        acc0 = wmma_bf16(a, b0, acc0);   // 3 independent chains: dependent
        acc1 = wmma_bf16(a, b1, acc1);   // WMMAs are >= 3 issue slots apart
        acc2 = wmma_bf16(a, b2, acc2);
      }
      store_qkv_tile(sm, acc0, t0, h, mt, crow, ccol, qkv_b);
      store_qkv_tile(sm, acc1, t1, h, mt, crow, ccol, qkv_b);
      store_qkv_tile(sm, acc2, t2, h, mt, crow, ccol, qkv_b);
    }
    __syncthreads();

    // ---- S = q * k^T (+bias+mask): 8 jobs = 4 mt x 2 ntile-pairs ----
    {
      const int mt = wave >> 1, np = wave & 1;
      v16bf a  = load_a16(&sm.q[0][0], mt * 16, HDIM, 0, lane);
      v16bf b0 = load_b16(&sm.k[0][0], (np * 2 + 0) * 16, HDIM, 0, lane);
      v16bf b1 = load_b16(&sm.k[0][0], (np * 2 + 1) * 16, HDIM, 0, lane);
      v8f acc0 = {}, acc1 = {};
      acc0 = wmma_bf16(a, b0, acc0);
      acc1 = wmma_bf16(a, b1, acc1);
#pragma unroll
      for (int t = 0; t < 2; ++t) {
        const v8f& acc = t ? acc1 : acc0;
        const int col = (np * 2 + t) * 16 + ccol;
#pragma unroll
        for (int i = 0; i < 8; ++i) {
          const int row = mt * 16 + crow + i;
          float sv;
          if (row < WIN_N && col < WIN_N)
            sv = acc[i] + biasf[(h * WIN_N + row) * WIN_N + col] +
                 mw[row * WIN_N + col];
          else
            sv = -1e30f;  // pad -> exp()==0
          sm.attn[row][col] = sv;
        }
      }
    }
    __syncthreads();

    // ---- row softmax: 4 lanes per row, all 256 threads active ----
    {
      const int row = tid >> 2;
      const int sub = tid & 3;
      float mx = -1e30f;
      for (int c = sub; c < WIN_N; c += 4) mx = fmaxf(mx, sm.attn[row][c]);
      mx = fmaxf(mx, __shfl_xor(mx, 1, 32));
      mx = fmaxf(mx, __shfl_xor(mx, 2, 32));
      float sum = 0.f;
      for (int c = sub; c < WIN_N; c += 4) sum += __expf(sm.attn[row][c] - mx);
      sum += __shfl_xor(sum, 1, 32);
      sum += __shfl_xor(sum, 2, 32);
      const float inv = 1.0f / sum;
      for (int c = sub; c < NPAD; c += 4)
        sm.p[row][c] = (c < WIN_N)
                           ? (bf16_t)(__expf(sm.attn[row][c] - mx) * inv)
                           : (bf16_t)0.0f;
    }
    __syncthreads();

    // ---- O_h = P * V: 8 jobs (4 mt x 2 nt); K split into two INDEPENDENT
    //      accumulators (no WMMA->WMMA RAW), summed in VALU at the end ----
    {
      const int mt = wave >> 1, nt = wave & 1;
      v16bf a0 = load_a16(&sm.p[0][0], mt * 16, NPAD, 0, lane);
      v16bf a1 = load_a16(&sm.p[0][0], mt * 16, NPAD, 32, lane);
      v16bf b0 = load_b16(&sm.vT[0][0], nt * 16, NPAD, 0, lane);
      v16bf b1 = load_b16(&sm.vT[0][0], nt * 16, NPAD, 32, lane);
      v8f acc0 = {}, acc1 = {};
      acc0 = wmma_bf16(a0, b0, acc0);
      acc1 = wmma_bf16(a1, b1, acc1);
#pragma unroll
      for (int i = 0; i < 8; ++i)
        sm.o[mt * 16 + crow + i][h * HDIM + nt * 16 + ccol] =
            (bf16_t)(acc0[i] + acc1[i]);
    }
    __syncthreads();
  }

  // ---- projection: out = O(49x192)*proj_w^T + proj_b, 16 triple jobs ----
#pragma unroll 1
  for (int j = wave; j < 16; j += 8) {
    const int mt = j >> 2, g = j & 3;
    const int c0 = g * 48;  // this job covers output cols [c0, c0+48)
    v8f acc0 = {}, acc1 = {}, acc2 = {};
#pragma unroll
    for (int kk = 0; kk < 6; ++kk) {
      v16bf a  = load_a16(&sm.o[0][0], mt * 16, NDIM, kk * 32, lane);
      v16bf b0 = load_b16(projw, c0,      NDIM, kk * 32, lane);
      v16bf b1 = load_b16(projw, c0 + 16, NDIM, kk * 32, lane);
      v16bf b2 = load_b16(projw, c0 + 32, NDIM, kk * 32, lane);
      acc0 = wmma_bf16(a, b0, acc0);
      acc1 = wmma_bf16(a, b1, acc1);
      acc2 = wmma_bf16(a, b2, acc2);
    }
    const float pb0 = proj_b[c0 + ccol];
    const float pb1 = proj_b[c0 + 16 + ccol];
    const float pb2 = proj_b[c0 + 32 + ccol];
#pragma unroll
    for (int i = 0; i < 8; ++i) {
      const int row = mt * 16 + crow + i;
      if (row < WIN_N) {
        float* orow = out + (size_t)b * WIN_N * NDIM + row * NDIM;
        orow[c0 + ccol]      = acc0[i] + pb0;
        orow[c0 + 16 + ccol] = acc1[i] + pb1;
        orow[c0 + 32 + ccol] = acc2[i] + pb2;
      }
    }
  }
}

extern "C" void kernel_launch(void* const* d_in, const int* in_sizes, int n_in,
                              void* d_out, int out_size, void* d_ws,
                              size_t ws_size, hipStream_t stream) {
  (void)in_sizes; (void)n_in; (void)out_size; (void)ws_size;
  const float* x      = (const float*)d_in[0];
  const float* mask   = (const float*)d_in[1];
  const float* qkv_w  = (const float*)d_in[2];
  const float* qkv_b  = (const float*)d_in[3];
  const float* proj_w = (const float*)d_in[4];
  const float* proj_b = (const float*)d_in[5];
  const float* rpb    = (const float*)d_in[6];
  const int*   reli   = (const int*)d_in[7];

  char*   ws    = (char*)d_ws;
  bf16_t* qkvw  = (bf16_t*)(ws);                       // 576*192*2 = 221184 B
  bf16_t* projw = (bf16_t*)(ws + 221184);              // 192*192*2 =  73728 B
  float*  biasf = (float*)(ws + 221184 + 73728);       // 6*49*49*4 =  57624 B

  const int prep_elems = 3 * NDIM * NDIM;  // largest conversion
  prep_kernel<<<(prep_elems + 255) / 256, 256, 0, stream>>>(
      qkv_w, proj_w, rpb, reli, qkvw, projw, biasf);

  win_attn_kernel<<<NBATCH, 256, 0, stream>>>(x, mask, qkv_b, proj_b, qkvw,
                                              projw, biasf, (float*)d_out);
}
